// MessagePassingLayer_39316130627980
// MI455X (gfx1250) — compile-verified, matrix operand order checked by reading, not run on hardware
//
#include <hip/hip_runtime.h>
#include <math.h>

// ---------------------------------------------------------------------------
// MessagePassingLayer fused kernel for gfx1250 (MI455X, wave32, WMMA + TDM).
// B=4, N=4096, D=256, E=32768, MSG=256.
//
// Fully fused per 16-edge-row wave tile:
//   gather pair -> bf16 WMMA GEMM1 + gelu -> LDS -> GEMM2 + gelu -> LDS
//   -> GEMM3 -> msg*sigmoid(gate) -> global_atomic_add_f32 scatter.
// Weights pre-converted once to bf16 (transposed [n][k]) in d_ws; weight
// panels staged to LDS by the Tensor Data Mover with hardware padding
// (64B data + 16B pad = 80B row stride), double-buffered so each DMA
// overlaps the previous panel's WMMA stream.
// ---------------------------------------------------------------------------

typedef __attribute__((ext_vector_type(16))) __bf16 v16bf;
typedef __attribute__((ext_vector_type(2)))  __bf16 v2bf;
typedef __attribute__((ext_vector_type(8)))  float  v8f;
typedef __attribute__((ext_vector_type(4)))  unsigned int u32x4;
typedef __attribute__((ext_vector_type(8)))  int          i32x8;
typedef __attribute__((ext_vector_type(4)))  int          i32x4;

constexpr int Bsz = 4, Nn = 4096, Dd = 256, Ee = 32768;
constexpr int BLOCK   = 128;   // 4 waves of 32
constexpr int WAVES   = 4;
constexpr int WG_ROWS = 64;    // 4 waves x 16 rows each
constexpr int PSTRIDE = 40;    // panel row stride in ushorts (32 data + 8 pad = 80B)
constexpr int PANEL_ELEMS = 256 * PSTRIDE;       // 10240 ushorts = 20480 B

// bf16 transposed weights in workspace (byte offsets)
constexpr size_t W1T_OFF = 0;                       // [256 n][512 k] bf16
constexpr size_t W2T_OFF = 262144;                  // [256 n][256 k] bf16
constexpr size_t W3T_OFF = 262144 + 131072;         // [512 n][256 k] bf16

#if __has_builtin(__builtin_amdgcn_tensor_load_to_lds)
#define HAVE_TDM 1
#else
#define HAVE_TDM 0
#endif

union Frag {
  v16bf bf;
  uint4 q[2];
  unsigned u[8];
};

__device__ __forceinline__ unsigned short f2bf(float f) {
  unsigned u = __float_as_uint(f);
  u += 0x7FFFu + ((u >> 16) & 1u);   // round-to-nearest-even
  return (unsigned short)(u >> 16);
}
#if __has_builtin(__builtin_amdgcn_cvt_pk_bf16_f32)
__device__ __forceinline__ unsigned cvtpk(float x, float y) {
  union { v2bf v; unsigned u; } c;
  c.v = __builtin_amdgcn_cvt_pk_bf16_f32(x, y);
  return c.u;
}
#else
__device__ __forceinline__ unsigned cvtpk(float x, float y) {
  return (unsigned)f2bf(x) | ((unsigned)f2bf(y) << 16);
}
#endif
__device__ __forceinline__ float gelu_exact(float x) {
  return 0.5f * x * (1.0f + erff(x * 0.70710678118654752f));
}
__device__ __forceinline__ float sigm(float x) { return 1.0f / (1.0f + __expf(-x)); }
__device__ __forceinline__ v8f splat8(float x) { v8f r = {x,x,x,x,x,x,x,x}; return r; }

__device__ __forceinline__ void pack8(Frag& f, int base, float4 lo, float4 hi) {
  f.u[base+0] = cvtpk(lo.x, lo.y);
  f.u[base+1] = cvtpk(lo.z, lo.w);
  f.u[base+2] = cvtpk(hi.x, hi.y);
  f.u[base+3] = cvtpk(hi.z, hi.w);
}

// ---------------------------------------------------------------------------
// TDM: DMA a [rows x 32] bf16 tile from a row-major [*, dim0] bf16 matrix
// into LDS with hardware row padding (64B data + 16B pad = 80B stride).
// Issued by one wave; completion via s_wait_tensorcnt. (ISA 08_async_tensor §8)
// ---------------------------------------------------------------------------
#if HAVE_TDM
__device__ __forceinline__ void tdm_load_panel(unsigned lds_off,
                                               const unsigned short* gptr,
                                               unsigned dim0, unsigned rows) {
  unsigned long long ga = (unsigned long long)(uintptr_t)gptr;
  u32x4 g0;
  g0[0] = 1u;                                            // count=1, no gather
  g0[1] = lds_off;                                       // lds_addr (bytes)
  g0[2] = (unsigned)(ga & 0xFFFFFFFFull);                // global_addr[31:0]
  g0[3] = (unsigned)((ga >> 32) & 0x01FFFFFFull)         // global_addr[56:32]
        | 0x80000000u;                                   // type=2 ("image")
  i32x8 g1;
  g1[0] = (int)((1u << 16)        // data_size = 2 bytes
              | (1u << 20)        // pad_enable
              | (3u << 22)        // pad_interval: 16 DWORDs (=64B) between pads
              | (3u << 25));      // pad_amount: 4 DWORDs (=16B)
  g1[1] = (int)((dim0 & 0xFFFFu) << 16);                              // dim0 lo
  g1[2] = (int)(((dim0 >> 16) & 0xFFFFu) | ((rows & 0xFFFFu) << 16)); // dim0 hi | dim1 lo
  g1[3] = (int)(((rows >> 16) & 0xFFFFu) | (32u << 16));              // dim1 hi | tile_dim0=32
  g1[4] = (int)(rows & 0xFFFFu);                                      // tile_dim1 | tile_dim2=0
  g1[5] = (int)dim0;                                                  // dim0_stride lo32
  g1[6] = 0;
  g1[7] = 0;
  i32x4 gz4 = {0, 0, 0, 0};
  i32x8 gz8 = {0, 0, 0, 0, 0, 0, 0, 0};
  __builtin_amdgcn_tensor_load_to_lds(g0, g1, gz4, gz4, gz8, 0);
}
#else
// Fallback: cooperative b128 copy (all BLOCK threads).
__device__ __forceinline__ void copy_panel(unsigned short* sp,
                                           const unsigned short* gptr,
                                           unsigned dim0, unsigned rows) {
  const int nchunk = (int)rows * 4;            // 4 x 16B per 64B row
  for (int ch = threadIdx.x; ch < nchunk; ch += BLOCK) {
    const int r = ch >> 2, c = ch & 3;
    uint4 v = *(const uint4*)(gptr + (size_t)r * dim0 + c * 8);
    *(uint4*)(sp + (size_t)r * PSTRIDE + c * 8) = v;
  }
}
#endif

// B fragment (K=32 x N=16 bf16) from an LDS panel row (layout mirrors A).
__device__ __forceinline__ Frag loadB(const unsigned short* sp, int prow, int kb) {
  Frag f;
  const unsigned short* p = sp + prow * PSTRIDE + kb;
  f.q[0] = *(const uint4*)(p);
  f.q[1] = *(const uint4*)(p + 16);
  return f;
}

// A fragment (M=16 x K=32 bf16) from the per-wave LDS activation scratch.
__device__ __forceinline__ Frag loadA_lds(const unsigned short* hp, int m, int k0, int kb) {
  Frag f;
  const unsigned short* p = hp + m * 256 + k0 + kb;
  f.q[0] = *(const uint4*)(p);
  f.q[1] = *(const uint4*)(p + 16);
  return f;
}

#define WMMA_BF16(A, Bm, C) \
  __builtin_amdgcn_wmma_f32_16x16x32_bf16(false, (A), false, (Bm), (short)0, (C), false, false)

__global__ __launch_bounds__(BLOCK, 1) void mp_fused(
    const float* __restrict__ nodes, const long long* __restrict__ edges,
    const float* __restrict__ b1, const float* __restrict__ b2,
    const float* __restrict__ b3,
    const unsigned short* __restrict__ wt,   // bf16 W1t/W2t/W3t in workspace
    float* __restrict__ out)
{
  __shared__ alignas(16) unsigned short s_panel[2 * PANEL_ELEMS];  // 40960 B, double buffer
  __shared__ alignas(16) unsigned short s_h[WAVES * 16 * 256];     // 32768 B
  __shared__ int s_dst[WG_ROWS];

  const unsigned short* W1t = wt + W1T_OFF / 2;   // [256][512]
  const unsigned short* W2t = wt + W2T_OFF / 2;   // [256][256]
  const unsigned short* W3t = wt + W3T_OFF / 2;   // [512][256]

  const int tid   = threadIdx.x;
  const int wave  = tid >> 5;
  const int lane  = tid & 31;
  const int nloc  = lane & 15;
  const int khalf = lane >> 4;
  const int kb    = khalf << 3;    // 0 or 8: lane's K sub-run base

  const int rowBase = blockIdx.x * WG_ROWS;      // block never straddles b
  const int b   = rowBase / Ee;
  const int row = rowBase + wave * 16 + nloc;    // this lane's edge row (M)
  const int e   = row - b * Ee;

  const long long srcn = edges[2LL * e];
  const long long dstn = edges[2LL * e + 1];
  if (lane < 16) s_dst[wave * 16 + nloc] = (int)dstn;

  const float* nb   = nodes + (size_t)b * Nn * Dd;
  const float* srcp = nb + (size_t)srcn * Dd;
  const float* dstp = nb + (size_t)dstn * Dd;
  unsigned short* hp = s_h + wave * (16 * 256);  // per-wave 16x256 bf16 scratch

#if HAVE_TDM
  const unsigned panel_lds = (unsigned)(uintptr_t)(&s_panel[0]);
#define PBASE(buf) (panel_lds + (unsigned)(buf) * (unsigned)(2 * PANEL_ELEMS))
#endif

  // ---------------- GEMM1: h1 = gelu(pair @ W1 + b1), K=512, N=256 --------
  v8f acc[16];
  #pragma unroll
  for (int nt = 0; nt < 16; ++nt) acc[nt] = splat8(b1[nt * 16 + nloc]);

#if HAVE_TDM
  if (wave == 0) tdm_load_panel(PBASE(0), W1t + 0, 512u, 256u);   // prologue
#endif
  for (int kt = 0; kt < 16; ++kt) {
    const int k0  = kt * 32;
    const int cur = kt & 1;
#if HAVE_TDM
    if (wave == 0) {
      if (kt < 15) {       // prefetch next panel into the other buffer
        tdm_load_panel(PBASE(1 - cur), W1t + k0 + 32, 512u, 256u);
        __builtin_amdgcn_s_wait_tensorcnt(1);   // panel kt landed; kt+1 in flight
      } else {
        __builtin_amdgcn_s_wait_tensorcnt(0);
      }
    }
#else
    copy_panel(s_panel + cur * PANEL_ELEMS, W1t + k0, 512u, 256u);
#endif
    __syncthreads();                     // panel kt visible to all waves
    const unsigned short* sp = s_panel + cur * PANEL_ELEMS;
    // Gather A fragment straight from the node rows (src half then dst half).
    const float* abase = (k0 < Dd) ? (srcp + k0) : (dstp + (k0 - Dd));
    Frag a;
    {
      float4 lo0 = *(const float4*)(abase + kb);
      float4 hi0 = *(const float4*)(abase + kb + 4);
      float4 lo1 = *(const float4*)(abase + kb + 16);
      float4 hi1 = *(const float4*)(abase + kb + 20);
      pack8(a, 0, lo0, hi0);
      pack8(a, 4, lo1, hi1);
    }
    #pragma unroll
    for (int nt = 0; nt < 16; ++nt) {
      Frag bm = loadB(sp, nt * 16 + nloc, kb);
      acc[nt] = WMMA_BF16(a.bf, bm.bf, acc[nt]);
    }
    __syncthreads();                     // compute kt done; buffer may be refilled
  }
#if HAVE_TDM
  if (wave == 0) tdm_load_panel(PBASE(0), W2t + 0, 256u, 256u);   // GEMM2 prologue (overlaps gelu)
#endif
  #pragma unroll
  for (int nt = 0; nt < 16; ++nt) {
    const int n = nt * 16 + nloc;
    #pragma unroll
    for (int v = 0; v < 8; ++v) {
      const int mm = v + 8 * khalf;
      hp[mm * 256 + n] = f2bf(gelu_exact(acc[nt][v]));
    }
  }

  // ---------------- GEMM2: h2 = gelu(h1 @ W2 + b2), K=256, N=256 ----------
  #pragma unroll
  for (int nt = 0; nt < 16; ++nt) acc[nt] = splat8(b2[nt * 16 + nloc]);
  for (int kt = 0; kt < 8; ++kt) {
    const int k0  = kt * 32;
    const int cur = kt & 1;
#if HAVE_TDM
    if (wave == 0) {
      if (kt < 7) {
        tdm_load_panel(PBASE(1 - cur), W2t + k0 + 32, 256u, 256u);
        __builtin_amdgcn_s_wait_tensorcnt(1);
      } else {
        __builtin_amdgcn_s_wait_tensorcnt(0);
      }
    }
#else
    copy_panel(s_panel + cur * PANEL_ELEMS, W2t + k0, 256u, 256u);
#endif
    __syncthreads();
    const unsigned short* sp = s_panel + cur * PANEL_ELEMS;
    Frag a = loadA_lds(hp, nloc, k0, kb);
    #pragma unroll
    for (int nt = 0; nt < 16; ++nt) {
      Frag bm = loadB(sp, nt * 16 + nloc, kb);
      acc[nt] = WMMA_BF16(a.bf, bm.bf, acc[nt]);
    }
    __syncthreads();
  }
#if HAVE_TDM
  if (wave == 0) {                       // GEMM3 h=0 prologue (overlaps gelu)
    tdm_load_panel(PBASE(0), W3t + 0, 256u, 128u);
    tdm_load_panel(PBASE(0) + 128u * PSTRIDE * 2u, W3t + (size_t)Dd * 256, 256u, 128u);
  }
#endif
  #pragma unroll
  for (int nt = 0; nt < 16; ++nt) {
    const int n = nt * 16 + nloc;
    #pragma unroll
    for (int v = 0; v < 8; ++v) {
      const int mm = v + 8 * khalf;
      hp[mm * 256 + n] = f2bf(gelu_exact(acc[nt][v]));   // overwrite with h2
    }
  }

  // ------ GEMM3: out = h2 @ W3 + b3 ; contrib = msg * sigmoid(gate) -------
  int drow[8];
  #pragma unroll
  for (int v = 0; v < 8; ++v) drow[v] = s_dst[wave * 16 + v + 8 * khalf];

  for (int h = 0; h < 2; ++h) {          // two 128-wide N halves
    v8f accM[8], accG[8];
    #pragma unroll
    for (int nt = 0; nt < 8; ++nt) {
      const int n = h * 128 + nt * 16 + nloc;
      accM[nt] = splat8(b3[n]);
      accG[nt] = splat8(b3[Dd + n]);     // gate columns are offset by 256
    }
    const unsigned short* Wmsg = W3t + (size_t)(h * 128) * 256;
    const unsigned short* Wgat = W3t + (size_t)(Dd + h * 128) * 256;
    for (int kt = 0; kt < 8; ++kt) {
      const int k0  = kt * 32;
      const int cur = kt & 1;
#if HAVE_TDM
      if (wave == 0) {
        if (kt < 7) {                    // prefetch next pair (msg rows + gate rows)
          tdm_load_panel(PBASE(1 - cur), Wmsg + k0 + 32, 256u, 128u);
          tdm_load_panel(PBASE(1 - cur) + 128u * PSTRIDE * 2u, Wgat + k0 + 32, 256u, 128u);
          __builtin_amdgcn_s_wait_tensorcnt(2);   // current pair landed
        } else {
          __builtin_amdgcn_s_wait_tensorcnt(0);
        }
      }
#else
      copy_panel(s_panel + cur * PANEL_ELEMS, Wmsg + k0, 256u, 128u);
      copy_panel(s_panel + cur * PANEL_ELEMS + (size_t)128 * PSTRIDE, Wgat + k0, 256u, 128u);
#endif
      __syncthreads();
      const unsigned short* sp = s_panel + cur * PANEL_ELEMS;
      Frag a = loadA_lds(hp, nloc, k0, kb);
      #pragma unroll
      for (int nt = 0; nt < 8; ++nt) {
        Frag bm = loadB(sp, nt * 16 + nloc, kb);
        accM[nt] = WMMA_BF16(a.bf, bm.bf, accM[nt]);
        Frag bg = loadB(sp, 128 + nt * 16 + nloc, kb);
        accG[nt] = WMMA_BF16(a.bf, bg.bf, accG[nt]);
      }
      __syncthreads();
    }
#if HAVE_TDM
    if (h == 0 && wave == 0) {           // h=1 prologue (overlaps h=0 scatter)
      tdm_load_panel(PBASE(0), W3t + (size_t)128 * 256, 256u, 128u);
      tdm_load_panel(PBASE(0) + 128u * PSTRIDE * 2u,
                     W3t + (size_t)(Dd + 128) * 256, 256u, 128u);
    }
#endif
    #pragma unroll
    for (int nt = 0; nt < 8; ++nt) {
      const int n = h * 128 + nt * 16 + nloc;
      #pragma unroll
      for (int v = 0; v < 8; ++v) {
        float c = accM[nt][v] * sigm(accG[nt][v]);
        float* dp = out + ((size_t)b * Nn + drow[v]) * Dd + n;
        unsafeAtomicAdd(dp, c);          // global_atomic_add_f32
      }
    }
  }
}

// ---------------------------------------------------------------------------
// One-time per launch: convert weights f32 -> bf16, transposed to [n][k].
// ---------------------------------------------------------------------------
__global__ void prep_weights(const float* __restrict__ W1,
                             const float* __restrict__ W2,
                             const float* __restrict__ W3,
                             unsigned short* __restrict__ wt) {
  const int i = blockIdx.x * blockDim.x + threadIdx.x;
  if (i < 131072) {                                   // W1t [256][512] <- W1 [512][256]
    const int n = i >> 9, k = i & 511;
    wt[W1T_OFF / 2 + i] = f2bf(W1[(size_t)k * 256 + n]);
  } else if (i < 131072 + 65536) {                    // W2t [256][256] <- W2 [256][256]
    const int j = i - 131072;
    const int n = j >> 8, k = j & 255;
    wt[W2T_OFF / 2 + j] = f2bf(W2[(size_t)k * 256 + n]);
  } else if (i < 131072 + 65536 + 131072) {           // W3t [512][256] <- W3 [256][512]
    const int j = i - 131072 - 65536;
    const int n = j >> 8, k = j & 255;
    wt[W3T_OFF / 2 + j] = f2bf(W3[(size_t)k * 512 + n]);
  }
}

// Zero the (poisoned) output before the scatter-add kernel.
__global__ void zero_f32(float* __restrict__ p, int n4) {
  int i = blockIdx.x * blockDim.x + threadIdx.x;
  if (i < n4) ((float4*)p)[i] = make_float4(0.f, 0.f, 0.f, 0.f);
}

extern "C" void kernel_launch(void* const* d_in, const int* in_sizes, int n_in,
                              void* d_out, int out_size, void* d_ws, size_t ws_size,
                              hipStream_t stream) {
  const float*     nodes = (const float*)d_in[0];
  const long long* edges = (const long long*)d_in[1];
  const float*     W1    = (const float*)d_in[2];
  const float*     b1    = (const float*)d_in[3];
  const float*     W2    = (const float*)d_in[4];
  const float*     b2    = (const float*)d_in[5];
  const float*     W3    = (const float*)d_in[6];
  const float*     b3    = (const float*)d_in[7];
  float* out = (float*)d_out;
  unsigned short* wt = (unsigned short*)d_ws;       // needs 640 KB of d_ws

  const int n_out = Bsz * Nn * Dd;                  // 4,194,304 floats
  zero_f32<<<(n_out / 4 + 255) / 256, 256, 0, stream>>>(out, n_out / 4);

  const int n_w = 131072 + 65536 + 131072;          // 327,680 bf16 elements
  prep_weights<<<(n_w + 255) / 256, 256, 0, stream>>>(W1, W2, W3, wt);

  const int n_rows = Bsz * Ee;                      // 131072 edge instances
  mp_fused<<<n_rows / WG_ROWS, BLOCK, 0, stream>>>(nodes, edges, b1, b2, b3, wt, out);
}
